// MemoryController_35648228557109
// MI455X (gfx1250) — compile-verified, matrix operand order checked by reading, not run on hardware
//
#include <hip/hip_runtime.h>
#include <hip/hip_bf16.h>
#include <stdint.h>

// Problem sizes (fixed by the reference)
#define B_ 64
#define T_ 128
#define I_ 2048
#define S_ 128
#define D_ 256

// LDS strides.
// DP=260 floats: rows are 1040 B (16-B aligned -> float4/b128 LDS ops);
//   260%64==4 -> worst case 2-way bank conflict on column walks.
// SA=260 ushorts: rows are 520 B (8-B aligned for d%4==0 -> b64 stores).
#define DP 260
#define SA 260

typedef __attribute__((ext_vector_type(16))) __bf16 bf16x16;
typedef __attribute__((ext_vector_type(8)))  float  f32x8;

union Frag16 {
  unsigned int u[8];
  uint4        q[2];
  bf16x16      v;
};

__device__ __forceinline__ unsigned short f2bf(float x) {
  unsigned int u = __float_as_uint(x);
  u += 0x7FFFu + ((u >> 16) & 1u);          // round-to-nearest-even
  return (unsigned short)(u >> 16);
}
__device__ __forceinline__ float bf2f(unsigned short h) {
  return __uint_as_float(((unsigned int)h) << 16);
}
__device__ __forceinline__ f32x8 wmma_bf16(const Frag16& a, const Frag16& b, f32x8 c) {
  return __builtin_amdgcn_wmma_f32_16x16x32_bf16(false, a.v, false, b.v, (short)0, c, false, false);
}

// A-fragment (16x32 bf16) gather: lane holds row M=lane&15; element j holds
// K = kb + 2j + (j>=4 ? 8:0), with kb already including (lane>=16)*8 and the
// k-step base. Pairs are contiguous bf16 -> one dword load each (the compiler
// coalesces these into b128 loads across unrolled k-steps).
__device__ __forceinline__ void load_a_frag(Frag16& a, const unsigned short* p) {
#pragma unroll
  for (int j = 0; j < 8; ++j) {
    const int k = 2 * j + ((j >= 4) ? 8 : 0);
    a.u[j] = *(const unsigned int*)(p + k);
  }
}

// ---------------------------------------------------------------------------
// Prep kernel 1: fp32 -> packed bf16 pairs (for hidden_states)
// ---------------------------------------------------------------------------
__global__ void cvt_f32_to_bf16_pairs(const float* __restrict__ x,
                                      unsigned int* __restrict__ out, long n2) {
  long i = (long)blockIdx.x * blockDim.x + threadIdx.x;
  const long stride = (long)gridDim.x * blockDim.x;
  for (; i < n2; i += stride) {
    const unsigned int lo = f2bf(x[2 * i]);
    const unsigned int hi = f2bf(x[2 * i + 1]);
    out[i] = lo | (hi << 16);
  }
}

// ---------------------------------------------------------------------------
// Prep kernel 2: swizzle W [K,256] fp32 row-major into WMMA B-fragment order.
// Block (n_tile,kstep) = 32 lanes x 8 dwords contiguous (1 KB) so a wave can
// fetch a fragment with two global_load_b128 per lane.
// ---------------------------------------------------------------------------
__global__ void swizzle_wmma_b(const float* __restrict__ W,
                               unsigned int* __restrict__ out, int K) {
  const int  N      = 256;
  const int  ksteps = K >> 5;
  const long total  = (long)K * N / 2;
  long i = (long)blockIdx.x * blockDim.x + threadIdx.x;
  const long stride = (long)gridDim.x * blockDim.x;
  for (; i < total; i += stride) {
    const int  j    = (int)(i & 7);
    const int  lane = (int)((i >> 3) & 31);
    const long blk  = i >> 8;
    const int  ks   = (int)(blk % ksteps);
    const int  nt   = (int)(blk / ksteps);
    const int  n    = nt * 16 + (lane & 15);
    const int  k    = ks * 32 + (lane >> 4) * 8 + 2 * j + ((j >= 4) ? 8 : 0);
    const unsigned int lo = f2bf(W[(long)k * N + n]);
    const unsigned int hi = f2bf(W[(long)(k + 1) * N + n]);
    out[i] = lo | (hi << 16);
  }
}

// ---------------------------------------------------------------------------
// Hoisted projections: MI = X@W_in + b_in, VALS = X@W_val + b_val.
// Time-independent -> one [8192,2048]x[2048,256] WMMA GEMM (x2 outputs),
// done once outside the sequential scan.
// ---------------------------------------------------------------------------
__global__ void __launch_bounds__(256) precompute_mi_vals(
    const unsigned short* __restrict__ Xbf,
    const unsigned int* __restrict__ WinB, const unsigned int* __restrict__ WvalB,
    const float* __restrict__ b_in, const float* __restrict__ b_val,
    float* __restrict__ MI, float* __restrict__ VA) {
  const int lane = threadIdx.x & 31;
  const int wave = threadIdx.x >> 5;
  const int half = lane >> 4;
  const int nr   = lane & 15;
  const int rowBase = blockIdx.x * 128 + wave * 16;
  const int mA      = rowBase + nr;

  for (int nt = 0; nt < 16; ++nt) {
    f32x8 ci = {};
    f32x8 cv = {};
    for (int ks = 0; ks < 64; ++ks) {
      Frag16 a;
      const int kb = ks * 32 + half * 8;
      load_a_frag(a, Xbf + (size_t)mA * I_ + kb);
      Frag16 bi, bv;
      const size_t off = ((size_t)(nt * 64 + ks) * 32 + lane) * 8;
      const uint4* pi = (const uint4*)(WinB + off);
      const uint4* pv = (const uint4*)(WvalB + off);
      bi.q[0] = pi[0]; bi.q[1] = pi[1];
      bv.q[0] = pv[0]; bv.q[1] = pv[1];
      ci = wmma_bf16(a, bi, ci);
      cv = wmma_bf16(a, bv, cv);
    }
#pragma unroll
    for (int v = 0; v < 8; ++v) {
      const int m = rowBase + v + half * 8;   // C layout: M = v + 8*half
      const int n = nt * 16 + nr;             //           N = lane & 15
      MI[(size_t)m * D_ + n] = ci[v] + b_in[n];
      VA[(size_t)m * D_ + n] = cv[v] + b_val[n];
    }
  }
}

// ---------------------------------------------------------------------------
// Persistent scan kernel: one workgroup (8 wave32) per batch element.
// mem[S,D] fp32 master lives in LDS for all 128 steps (CDNA5 320KB LDS).
// Per step: sim + softmax (VALU), r/z gate GEMMs, rebuild A2=[vals|r*mem],
// h gate GEMM, blended update + per-slot L2 normalize.
// ---------------------------------------------------------------------------
#define SM_MEMF  0
#define SM_BUFA  (SM_MEMF + S_ * DP * 4)        // bf16(mem) half of A
#define SM_BUFB  (SM_BUFA + S_ * SA * 2)        // bf16(r*mem) half of A2
#define SM_MI    (SM_BUFB + S_ * SA * 2)
#define SM_VALS  (SM_MI   + D_ * 4)
#define SM_VBF   (SM_VALS + D_ * 4)
#define SM_SC    (SM_VBF  + D_ * 2)
#define SM_EW    (SM_SC   + S_ * 4)
#define SM_AARR  (SM_EW   + S_ * 4)
#define SM_USE   (SM_AARR + S_ * 4)
#define SM_TOTAL (SM_USE  + S_ * 4)             // = 270,848 B < 320 KB

__global__ void __launch_bounds__(256, 1) scan_kernel(
    const float* __restrict__ MEM0,
    const float* __restrict__ MI, const float* __restrict__ VA,
    const unsigned int* __restrict__ WrB, const unsigned int* __restrict__ WzB,
    const unsigned int* __restrict__ WhB,
    const float* __restrict__ b_r, const float* __restrict__ b_z,
    const float* __restrict__ b_h,
    unsigned short* __restrict__ ZG, float* __restrict__ OUT) {
  extern __shared__ unsigned char smem[];
  float*          memf    = (float*)(smem + SM_MEMF);
  unsigned short* bufA    = (unsigned short*)(smem + SM_BUFA);
  unsigned short* bufB    = (unsigned short*)(smem + SM_BUFB);
  float*          mi_s    = (float*)(smem + SM_MI);
  float*          valsf   = (float*)(smem + SM_VALS);
  unsigned short* vals_bf = (unsigned short*)(smem + SM_VBF);
  float*          scores  = (float*)(smem + SM_SC);
  float*          ew      = (float*)(smem + SM_EW);
  float*          aarr    = (float*)(smem + SM_AARR);
  float*          usage   = (float*)(smem + SM_USE);

  const int tid  = threadIdx.x;
  const int b    = blockIdx.x;
  const int lane = tid & 31;
  const int wave = tid >> 5;
  const int half = lane >> 4;
  const int nr   = lane & 15;
  const int mt   = wave;                        // wave -> slot-row tile

  // init state (vectorized: rows are 16-B aligned with DP=260)
  for (int e = tid * 4; e < S_ * D_; e += 256 * 4) {
    const int s = e >> 8, d = e & 255;
    *(float4*)(memf + (size_t)s * DP + d) = *(const float4*)(MEM0 + (size_t)b * S_ * D_ + e);
  }
  if (tid < S_) usage[tid] = 0.f;
  float age = 0.f;
  __syncthreads();

  unsigned short* zrow = ZG + (size_t)b * S_ * D_;

  for (int t = 0; t < T_; ++t) {
    const size_t row = (size_t)b * T_ + t;
    // load per-step vectors; stage bf16(mem) as WMMA-A mem-half (b128/b64 ops)
    {
      mi_s[tid]    = MI[row * D_ + tid];
      const float vv = VA[row * D_ + tid];
      valsf[tid]   = vv;
      vals_bf[tid] = f2bf(vv);
      for (int e = tid * 4; e < S_ * D_; e += 256 * 4) {
        const int s = e >> 8, d = e & 255;
        const float4 x = *(const float4*)(memf + (size_t)s * DP + d);
        uint2 p;
        p.x = (unsigned int)f2bf(x.x) | ((unsigned int)f2bf(x.y) << 16);
        p.y = (unsigned int)f2bf(x.z) | ((unsigned int)f2bf(x.w) << 16);
        *(uint2*)(bufA + (size_t)s * SA + d) = p;
      }
    }
    __syncthreads();

    // sim + scores (negated for softmax(-scores)); float4 LDS reads
    if (tid < S_) {
      const float4* mp4 = (const float4*)(memf + (size_t)tid * DP);
      const float4* mi4 = (const float4*)mi_s;
      float acc = 0.f;
      for (int d4 = 0; d4 < D_ / 4; ++d4) {
        const float4 a = mi4[d4];
        const float4 m = mp4[d4];
        acc = fmaf(a.x, m.x, acc);
        acc = fmaf(a.y, m.y, acc);
        acc = fmaf(a.z, m.z, acc);
        acc = fmaf(a.w, m.w, acc);
      }
      scores[tid] = -(acc - age * 0.1f - usage[tid] * 0.2f);
    }
    __syncthreads();
    if (tid < S_) {
      float mx = -3.4e38f;
      for (int s = 0; s < S_; ++s) mx = fmaxf(mx, scores[s]);   // broadcast reads
      ew[tid] = __expf(scores[tid] - mx);
    }
    __syncthreads();
    if (tid < S_) {
      float sum = 0.f;
      for (int s = 0; s < S_; ++s) sum += ew[s];
      const float w = ew[tid] / sum;
      const float msk = (w > 0.01f) ? 1.f : 0.f;
      aarr[tid]  = msk * w * 0.5f;                 // UPDATE_RATE = 0.5
      usage[tid] = (usage[tid] + msk * w) * 0.99f; // only read next step
    }
    __syncthreads();

    // r/z gate GEMMs: C[128x256] = [vals|mem][128x512] @ W[512x256]
    {
      const unsigned short* arow = bufA + (size_t)(mt * 16 + nr) * SA;
      for (int nt = 0; nt < 16; ++nt) {
        // hide L2 latency for the next n-tile's fragments (global_prefetch_b8)
        if (nt < 15) {
          const size_t noff = ((size_t)((nt + 1) * 16) * 32 + lane) * 8;
          __builtin_prefetch(WrB + noff, 0, 1);
          __builtin_prefetch(WzB + noff, 0, 1);
        }
        f32x8 cr = {};
        f32x8 cz = {};
        for (int ks = 0; ks < 16; ++ks) {
          Frag16 a;
          const int kb = (ks & 7) * 32 + half * 8;
          if (ks < 8) load_a_frag(a, vals_bf + kb);   // broadcast-row half
          else        load_a_frag(a, arow + kb);      // mem half
          Frag16 fr, fz;
          const size_t off = ((size_t)(nt * 16 + ks) * 32 + lane) * 8;
          const uint4* pr = (const uint4*)(WrB + off);
          const uint4* pz = (const uint4*)(WzB + off);
          fr.q[0] = pr[0]; fr.q[1] = pr[1];
          fz.q[0] = pz[0]; fz.q[1] = pz[1];
          cr = wmma_bf16(a, fr, cr);
          cz = wmma_bf16(a, fz, cz);
        }
#pragma unroll
        for (int v = 0; v < 8; ++v) {
          const int m = mt * 16 + v + half * 8;
          const int n = nt * 16 + nr;
          const float r = 1.f / (1.f + __expf(-(cr[v] + b_r[n])));
          const float z = 1.f / (1.f + __expf(-(cz[v] + b_z[n])));
          bufB[(size_t)m * SA + n] = f2bf(r * memf[(size_t)m * DP + n]);
          zrow[m * D_ + n] = f2bf(aarr[m] * z);   // a*z; same lane re-reads below
        }
      }
    }
    __syncthreads();

    // h gate GEMM with A2 = [vals | r*mem], then fused blended update:
    // mem' = mem*(1 - a*z) + (a*z)*tanh(...)
    {
      const unsigned short* arow = bufB + (size_t)(mt * 16 + nr) * SA;
      for (int nt = 0; nt < 16; ++nt) {
        if (nt < 15) {
          const size_t noff = ((size_t)((nt + 1) * 16) * 32 + lane) * 8;
          __builtin_prefetch(WhB + noff, 0, 1);
        }
        f32x8 ch = {};
        for (int ks = 0; ks < 16; ++ks) {
          Frag16 a;
          const int kb = (ks & 7) * 32 + half * 8;
          if (ks < 8) load_a_frag(a, vals_bf + kb);
          else        load_a_frag(a, arow + kb);
          Frag16 fh;
          const size_t off = ((size_t)(nt * 16 + ks) * 32 + lane) * 8;
          const uint4* ph = (const uint4*)(WhB + off);
          fh.q[0] = ph[0]; fh.q[1] = ph[1];
          ch = wmma_bf16(a, fh, ch);
        }
#pragma unroll
        for (int v = 0; v < 8; ++v) {
          const int m = mt * 16 + v + half * 8;
          const int n = nt * 16 + nr;
          const float cand = tanhf(ch[v] + b_h[n]);
          const float az   = bf2f(zrow[m * D_ + n]);
          const float old  = memf[(size_t)m * DP + n];
          memf[(size_t)m * DP + n] = old * (1.f - az) + az * cand;
        }
      }
    }
    __syncthreads();

    // per-slot L2 normalize (float4 LDS ops)
    if (tid < S_) {
      float4* rp4 = (float4*)(memf + (size_t)tid * DP);
      float ss = 0.f;
      for (int d4 = 0; d4 < D_ / 4; ++d4) {
        const float4 x = rp4[d4];
        ss = fmaf(x.x, x.x, ss);
        ss = fmaf(x.y, x.y, ss);
        ss = fmaf(x.z, x.z, ss);
        ss = fmaf(x.w, x.w, ss);
      }
      const float sc = 1.f / fmaxf(sqrtf(ss), 1e-12f);
      for (int d4 = 0; d4 < D_ / 4; ++d4) {
        float4 x = rp4[d4];
        x.x *= sc; x.y *= sc; x.z *= sc; x.w *= sc;
        rp4[d4] = x;
      }
    }
    age = age * 0.98f + 1.f;
    __syncthreads();
  }

  for (int e = tid * 4; e < S_ * D_; e += 256 * 4) {
    const int s = e >> 8, d = e & 255;
    *(float4*)(OUT + (size_t)b * S_ * D_ + e) = *(const float4*)(memf + (size_t)s * DP + d);
  }
}

// ---------------------------------------------------------------------------
// Host launcher
// ---------------------------------------------------------------------------
extern "C" void kernel_launch(void* const* d_in, const int* in_sizes, int n_in,
                              void* d_out, int out_size, void* d_ws, size_t ws_size,
                              hipStream_t stream) {
  (void)in_sizes; (void)n_in; (void)out_size; (void)ws_size;
  const float* X     = (const float*)d_in[0];
  const float* MEM0  = (const float*)d_in[1];
  const float* W_in  = (const float*)d_in[2];
  const float* b_in  = (const float*)d_in[3];
  const float* W_val = (const float*)d_in[4];
  const float* b_val = (const float*)d_in[5];
  const float* W_r   = (const float*)d_in[6];
  const float* b_r   = (const float*)d_in[7];
  const float* W_z   = (const float*)d_in[8];
  const float* b_z   = (const float*)d_in[9];
  const float* W_h   = (const float*)d_in[10];
  const float* b_h   = (const float*)d_in[11];

  // workspace carve-up (~54.75 MB total)
  char*  ws = (char*)d_ws;
  size_t o  = 0;
  auto carve = [&](size_t bytes) {
    void* p = ws + o;
    o += (bytes + 255) & ~(size_t)255;
    return p;
  };
  unsigned short* Xbf  = (unsigned short*)carve((size_t)B_ * T_ * I_ * 2); // 32 MB
  unsigned int*   WinB = (unsigned int*)carve((size_t)I_ * D_ * 2);       //  1 MB
  unsigned int*   WvaB = (unsigned int*)carve((size_t)I_ * D_ * 2);       //  1 MB
  unsigned int*   WrB  = (unsigned int*)carve((size_t)2 * D_ * D_ * 2);   // 256 KB
  unsigned int*   WzB  = (unsigned int*)carve((size_t)2 * D_ * D_ * 2);
  unsigned int*   WhB  = (unsigned int*)carve((size_t)2 * D_ * D_ * 2);
  float*          MIb  = (float*)carve((size_t)B_ * T_ * D_ * 4);         //  8 MB
  float*          VAb  = (float*)carve((size_t)B_ * T_ * D_ * 4);         //  8 MB
  unsigned short* ZG   = (unsigned short*)carve((size_t)B_ * S_ * D_ * 2);//  4 MB

  // 1) convert hidden_states to packed bf16
  const long n2 = (long)B_ * T_ * I_ / 2;
  cvt_f32_to_bf16_pairs<<<2048, 256, 0, stream>>>(X, (unsigned int*)Xbf, n2);

  // 2) swizzle all weight matrices into WMMA B-fragment layout
  swizzle_wmma_b<<<1024, 256, 0, stream>>>(W_in,  WinB, I_);
  swizzle_wmma_b<<<1024, 256, 0, stream>>>(W_val, WvaB, I_);
  swizzle_wmma_b<<< 256, 256, 0, stream>>>(W_r,   WrB,  2 * D_);
  swizzle_wmma_b<<< 256, 256, 0, stream>>>(W_z,   WzB,  2 * D_);
  swizzle_wmma_b<<< 256, 256, 0, stream>>>(W_h,   WhB,  2 * D_);

  // 3) hoisted input projections (time-independent)
  precompute_mi_vals<<<(B_ * T_) / 128, 256, 0, stream>>>(
      Xbf, WinB, WvaB, b_in, b_val, MIb, VAb);

  // 4) persistent per-batch scan (~271 KB dynamic LDS per workgroup)
  hipFuncSetAttribute((const void*)scan_kernel,
                      hipFuncAttributeMaxDynamicSharedMemorySize, SM_TOTAL);
  scan_kernel<<<B_, 256, SM_TOTAL, stream>>>(
      MEM0, MIb, VAb, WrB, WzB, WhB, b_r, b_z, b_h, ZG, (float*)d_out);
}
